// Attention_39453569581229
// MI455X (gfx1250) — matrix operand order repaired and checked by reading, not compiled, not run on hardware
//
#include <hip/hip_runtime.h>
#include <hip/hip_bf16.h>

typedef __attribute__((ext_vector_type(16))) __bf16        v16bf;
typedef __attribute__((ext_vector_type(8)))  float         v8f;
typedef __attribute__((ext_vector_type(4)))  unsigned int  u32x4;
typedef __attribute__((ext_vector_type(8)))  int           i32x8;
typedef __attribute__((ext_vector_type(4)))  int           i32x4;

#define D_MODEL 1024
#define SEQ_L   2048
#define NHEADS  16
#define HDIM    64
#define BATCH   4

#if defined(__has_builtin)
#if __has_builtin(__builtin_amdgcn_tensor_load_to_lds)
#define HAVE_TDM 1
#endif
#endif
#ifndef HAVE_TDM
#define HAVE_TDM 0
#endif

static __device__ __forceinline__ unsigned short f2bf(float f) {
    unsigned u = __float_as_uint(f);
    u += 0x7FFFu + ((u >> 16) & 1u);   // round-to-nearest-even
    return (unsigned short)(u >> 16);
}

union FragBF { v16bf v; u32x4 u[2]; };

// Load one 16x16x32 bf16 WMMA fragment as two 16-byte chunks (b128 each).
static __device__ __forceinline__ v16bf load_frag(const unsigned short* p0,
                                                  const unsigned short* p1) {
    FragBF f;
    f.u[0] = *(const u32x4*)p0;
    f.u[1] = *(const u32x4*)p1;
    return f.v;
}

static __device__ __forceinline__ v8f vzero8() {
    v8f z = {0.f,0.f,0.f,0.f,0.f,0.f,0.f,0.f};
    return z;
}

// ---- CDNA5 async global->LDS copy (ASYNCcnt) -------------------------------
// lds_off: workgroup-relative LDS byte offset (low 32 bits of a flat LDS ptr,
// per the flat LDS aperture mapping).
static __device__ __forceinline__ void async_ld_b128(unsigned lds_off,
                                                     const void* gaddr) {
    asm volatile("global_load_async_to_lds_b128 %0, %1, off"
                 :: "v"(lds_off), "v"(gaddr) : "memory");
}

static __device__ __forceinline__ void wait_async0() {
#if defined(__has_builtin) && __has_builtin(__builtin_amdgcn_s_wait_asynccnt)
    __builtin_amdgcn_s_wait_asynccnt(0);
#else
    asm volatile("s_wait_asynccnt 0" ::: "memory");
#endif
}

static __device__ __forceinline__ unsigned lds_off_of(const void* p) {
    return (unsigned)(uintptr_t)p;
}

// ---- CDNA5 Tensor Data Mover: 2-D tile load --------------------------------
#if HAVE_TDM
// flags_w0: group1 word0 = {pad_amount[31:25], pad_interval[24:22],
//                           early_timeout, pad_enable, iterate, atomic_bar,
//                           data_size[17:16], workgroup_mask[15:0]}
#define TDM_FLAGS_BF16        ((1u << 16))                                   // 2B elems
#define TDM_FLAGS_BF16_PAD40  ((1u << 16) | (1u << 20) | (3u << 22) | (3u << 25))
// pad: after every 16 DWORDs (64B row) insert 4 DWORDs (16B) -> 80B row stride

static __device__ __forceinline__ void tdm_load_2d(
    const unsigned short* gbase, unsigned lds_addr,
    unsigned tensor_d0, unsigned tensor_d1,
    unsigned tile_d0, unsigned tile_d1,
    unsigned long long stride_d0_elems, unsigned flags_w0)
{
    unsigned long long ga = (unsigned long long)(uintptr_t)gbase;
    u32x4 g0;
    g0[0] = 1u;                                                  // count=1 (valid)
    g0[1] = lds_addr;                                            // lds_addr[63:32]
    g0[2] = (unsigned)(ga & 0xFFFFFFFFull);                      // global_addr lo
    g0[3] = (unsigned)((ga >> 32) & 0x1FFFFFFull) | (2u << 30);  // addr hi | type=2
    i32x8 g1;
    g1[0] = (int)flags_w0;
    g1[1] = (int)((tensor_d0 & 0xFFFFu) << 16);                  // tensor_dim0 lo16
    g1[2] = (int)((tensor_d0 >> 16) | ((tensor_d1 & 0xFFFFu) << 16));
    g1[3] = (int)((tensor_d1 >> 16) | (tile_d0 << 16));          // | tile_dim0
    g1[4] = (int)(tile_d1 & 0xFFFFu);                            // tile_dim1, dim2=0
    g1[5] = (int)(stride_d0_elems & 0xFFFFFFFFull);              // dim0 stride lo32
    g1[6] = (int)((stride_d0_elems >> 32) & 0xFFFFull);          // stride hi16
    g1[7] = 0;
    i32x4 gz4 = {0, 0, 0, 0};
    i32x8 gz8 = {0, 0, 0, 0, 0, 0, 0, 0};
    // 6-arg form (clang-23 / therock headers): groups 0..3 + aux group + cpol
    __builtin_amdgcn_tensor_load_to_lds(g0, g1, gz4, gz4, gz8, 0);
}
#endif

// ---------------------------------------------------------------------------
// f32 -> bf16 conversion (grid-stride)
// ---------------------------------------------------------------------------
__global__ void cvt_f32_bf16(const float* __restrict__ in,
                             unsigned short* __restrict__ out, int n) {
    int i = blockIdx.x * blockDim.x + threadIdx.x;
    int stride = gridDim.x * blockDim.x;
    for (; i < n; i += stride) out[i] = f2bf(in[i]);
}

// ---------------------------------------------------------------------------
// bf16 GEMM: C = A(MxK) * Bw(NxK)^T + bias.
// MODE 0: QKV epilogue -> RoPE(q,k), write Qb/Kb row-major + Vt transposed
// MODE 1: out-projection epilogue -> f32 store to Cout
// Block: 256 thr (8 waves), 128x128 C tile; wave tile 32(M) x 64(N).
// A tile staged by TDM (wave 0), B tile by async global->LDS copies.
// ---------------------------------------------------------------------------
template <int MODE>
__global__ __launch_bounds__(256) void gemm_bf16_wmma(
    const unsigned short* __restrict__ A,
    const unsigned short* __restrict__ Bw,
    const float* __restrict__ bias,
    int M, int N, int K,
    unsigned short* __restrict__ Qb,
    unsigned short* __restrict__ Kb,
    unsigned short* __restrict__ Vt,
    float* __restrict__ Cout)
{
    __shared__ unsigned short As[128 * 40];   // 128 rows x 32 K, stride 40 (80B)
    __shared__ unsigned short Bs[128 * 40];

    const int tid   = threadIdx.x;
    const int lane  = tid & 31;
    const int wid   = tid >> 5;
    const int lhalf = lane >> 4;
    const int lm    = lane & 15;
    const int wm    = wid & 3;   // 4 waves along M
    const int wn    = wid >> 2;  // 2 waves along N
    const int mblk  = blockIdx.y * 128;
    const int nblk  = blockIdx.x * 128;

    v8f acc[2][4];
    #pragma unroll
    for (int a = 0; a < 2; ++a)
        #pragma unroll
        for (int b = 0; b < 4; ++b) acc[a][b] = vzero8();

    for (int k0 = 0; k0 < K; k0 += 32) {
        // ---- stage A tile: TDM descriptor (wave 0) or async fallback ----
#if HAVE_TDM
        if (wid == 0) {
            tdm_load_2d(&A[(size_t)mblk * K + k0], lds_off_of(As),
                        (unsigned)K, (unsigned)M, 32u, 128u,
                        (unsigned long long)K, TDM_FLAGS_BF16_PAD40);
        }
#else
        #pragma unroll
        for (int c = 0; c < 2; ++c) {
            int chunk = tid * 2 + c;
            int row   = chunk >> 2;
            int ko    = (chunk & 3) * 8;
            async_ld_b128(lds_off_of(&As[row * 40 + ko]),
                          &A[(size_t)(mblk + row) * K + k0 + ko]);
        }
#endif
        // ---- stage B tile: async global->LDS b128 copies ----
        #pragma unroll
        for (int c = 0; c < 2; ++c) {
            int chunk = tid * 2 + c;
            int row   = chunk >> 2;
            int ko    = (chunk & 3) * 8;
            async_ld_b128(lds_off_of(&Bs[row * 40 + ko]),
                          &Bw[(size_t)(nblk + row) * K + k0 + ko]);
        }
        if (k0 + 32 < K) {
            __builtin_prefetch(&A[(size_t)(mblk + tid) * K + k0 + 32], 0, 1);
            __builtin_prefetch(&Bw[(size_t)(nblk + (tid >> 1)) * K + k0 + 32], 0, 1);
        }
        wait_async0();
#if HAVE_TDM
        if (wid == 0) __builtin_amdgcn_s_wait_tensorcnt(0);
#endif
        __syncthreads();

        v16bf af[2], bf[4];
        #pragma unroll
        for (int mt = 0; mt < 2; ++mt) {
            const unsigned short* base = &As[(wm * 32 + mt * 16 + lm) * 40];
            int ko = lhalf ? 8 : 0;          // A layout: K 0-7/16-23 vs 8-15/24-31
            af[mt] = load_frag(base + ko, base + ko + 16);
        }
        #pragma unroll
        for (int nt = 0; nt < 4; ++nt) {
            const unsigned short* base = &Bs[(wn * 64 + nt * 16 + lm) * 40];
            int ko = lhalf ? 8 : 0;          // B layout: K 0-7/16-23 vs 8-15/24-31
            bf[nt] = load_frag(base + ko, base + ko + 16);
        }
        #pragma unroll
        for (int mt = 0; mt < 2; ++mt)
            #pragma unroll
            for (int nt = 0; nt < 4; ++nt)
                acc[mt][nt] = __builtin_amdgcn_wmma_f32_16x16x32_bf16(
                    false, af[mt], false, bf[nt], (short)0, acc[mt][nt], false, false);
        __syncthreads();
    }

    const int nw = nblk + wn * 64;           // wave's first output column
    float bv[4];
    #pragma unroll
    for (int nt = 0; nt < 4; ++nt) bv[nt] = bias[nw + nt * 16 + lm];

    if (MODE == 0) {
        const int sec = nw >> 10;            // 0=q, 1=k, 2=v
        const int rem = nw & 1023;
        const int h   = rem >> 6;
        #pragma unroll
        for (int mt = 0; mt < 2; ++mt) {
            #pragma unroll
            for (int i = 0; i < 8; ++i) {
                int Mrow = mblk + wm * 32 + mt * 16 + i + lhalf * 8;
                int bidx = Mrow >> 11;       // / SEQ_L
                int l    = Mrow & (SEQ_L - 1);
                size_t bh = (size_t)bidx * NHEADS + h;
                if (sec < 2) {
                    unsigned short* dst =
                        (sec == 0 ? Qb : Kb) + (bh * SEQ_L + l) * HDIM;
                    #pragma unroll
                    for (int jt = 0; jt < 2; ++jt) {
                        int d = jt * 16 + lm;                    // 0..31
                        float invf = __expf(-(float)d * (9.210340371976184f / 32.0f));
                        float ang  = (float)l * invf;
                        float cs = __cosf(ang), sn = __sinf(ang);
                        float x1 = acc[mt][jt][i]     + bv[jt];
                        float x2 = acc[mt][jt + 2][i] + bv[jt + 2];
                        dst[d]      = f2bf(x1 * cs - x2 * sn);
                        dst[d + 32] = f2bf(x2 * cs + x1 * sn);
                    }
                } else {
                    #pragma unroll
                    for (int jt = 0; jt < 4; ++jt) {
                        int d = jt * 16 + lm;
                        Vt[(bh * HDIM + d) * SEQ_L + l] = f2bf(acc[mt][jt][i] + bv[jt]);
                    }
                }
            }
        }
    } else {
        #pragma unroll
        for (int mt = 0; mt < 2; ++mt)
            #pragma unroll
            for (int i = 0; i < 8; ++i) {
                int Mrow = mblk + wm * 32 + mt * 16 + i + lhalf * 8;
                #pragma unroll
                for (int nt = 0; nt < 4; ++nt)
                    Cout[(size_t)Mrow * N + nw + nt * 16 + lm] = acc[mt][nt][i] + bv[nt];
            }
    }
}

// ---------------------------------------------------------------------------
// Flash attention: one wave handles a 16-query tile of one (b,h); loops over
// 64-key blocks with online softmax. K/V tiles staged into LDS shared by the
// whole workgroup: K via async copies, V via TDM (strided 2-D tile).
// Qb/Kb: [bh][L][64] bf16; Vt: [bh][64][L] bf16.
// ---------------------------------------------------------------------------
__global__ __launch_bounds__(256) void attn_flash_wmma(
    const unsigned short* __restrict__ Qb,
    const unsigned short* __restrict__ Kb,
    const unsigned short* __restrict__ Vt,
    unsigned short* __restrict__ Ob)
{
    __shared__ unsigned short Kt[64 * 64];        // keys x hd
    __shared__ unsigned short Vs[64 * 64];        // hd x keys (transposed V)
    __shared__ unsigned short Pst[8 * 16 * 64];   // per-wave P staging (C->A)

    const int tid   = threadIdx.x;
    const int lane  = tid & 31;
    const int wid   = tid >> 5;
    const int lhalf = lane >> 4;
    const int lm    = lane & 15;
    const int bh    = blockIdx.y;
    const int b     = bh >> 4;
    const int h     = bh & 15;
    const int q0    = blockIdx.x * 128 + wid * 16;

    const unsigned short* Qp = Qb + (size_t)bh * SEQ_L * HDIM;
    const unsigned short* Kp = Kb + (size_t)bh * SEQ_L * HDIM;
    const unsigned short* Vp = Vt + (size_t)bh * HDIM * SEQ_L;
    unsigned short* Pw = &Pst[wid * 16 * 64];

    // Q fragments (held in registers the whole loop)
    v16bf qf[2];
    #pragma unroll
    for (int s = 0; s < 2; ++s) {
        const unsigned short* base =
            Qp + (size_t)(q0 + lm) * HDIM + s * 32 + (lhalf ? 8 : 0);
        qf[s] = load_frag(base, base + 16);
    }

    float mi[8], li[8];
    v8f oacc[4];
    #pragma unroll
    for (int i = 0; i < 8; ++i) { mi[i] = -1e30f; li[i] = 0.f; }
    #pragma unroll
    for (int nt = 0; nt < 4; ++nt) oacc[nt] = vzero8();

    for (int kb = 0; kb < SEQ_L; kb += 64) {
        __syncthreads();   // all waves done reading previous tiles

        // stage K tile (contiguous 8KB) via async copies: 512 b128 chunks
        #pragma unroll
        for (int c = 0; c < 2; ++c) {
            int chunk = tid * 2 + c;
            int r  = chunk >> 3;
            int co = (chunk & 7) * 8;
            async_ld_b128(lds_off_of(&Kt[r * 64 + co]),
                          &Kp[(size_t)(kb + r) * HDIM + co]);
        }
        // stage V tile (64 rows of hd, strided by L) via TDM or async
#if HAVE_TDM
        if (wid == 0) {
            tdm_load_2d(Vp + kb, lds_off_of(Vs),
                        (unsigned)SEQ_L, (unsigned)HDIM, 64u, 64u,
                        (unsigned long long)SEQ_L, TDM_FLAGS_BF16);
        }
#else
        #pragma unroll
        for (int c = 0; c < 2; ++c) {
            int chunk = tid * 2 + c;
            int r  = chunk >> 3;
            int co = (chunk & 7) * 8;
            async_ld_b128(lds_off_of(&Vs[r * 64 + co]),
                          &Vp[(size_t)r * SEQ_L + kb + co]);
        }
#endif
        wait_async0();
#if HAVE_TDM
        if (wid == 0) __builtin_amdgcn_s_wait_tensorcnt(0);
#endif
        __syncthreads();

        // S = Q * K^T  (16 x 64 tile), K fragments from LDS
        v8f sacc[4];
        #pragma unroll
        for (int nt = 0; nt < 4; ++nt) sacc[nt] = vzero8();
        #pragma unroll
        for (int nt = 0; nt < 4; ++nt) {
            const unsigned short* krow = &Kt[(nt * 16 + lm) * 64];
            #pragma unroll
            for (int s = 0; s < 2; ++s) {
                const unsigned short* base = krow + s * 32 + (lhalf ? 8 : 0);
                v16bf kf = load_frag(base, base + 16);
                sacc[nt] = __builtin_amdgcn_wmma_f32_16x16x32_bf16(
                    false, qf[s], false, kf, (short)0, sacc[nt], false, false);
            }
        }

        // online softmax (rows live across 16 lanes per half-wave)
        #pragma unroll
        for (int i = 0; i < 8; ++i) {
            float mx = -1e30f;
            #pragma unroll
            for (int nt = 0; nt < 4; ++nt) {
                sacc[nt][i] *= 0.125f;                 // hd^-0.5
                mx = fmaxf(mx, sacc[nt][i]);
            }
            #pragma unroll
            for (int off = 1; off < 16; off <<= 1)
                mx = fmaxf(mx, __shfl_xor(mx, off, 32));
            float mnew  = fmaxf(mi[i], mx);
            float alpha = __expf(mi[i] - mnew);
            mi[i] = mnew;
            float rs = 0.f;
            #pragma unroll
            for (int nt = 0; nt < 4; ++nt) {
                float p = __expf(sacc[nt][i] - mnew);
                rs += p;
                Pw[(i + lhalf * 8) * 64 + nt * 16 + lm] = f2bf(p);
            }
            #pragma unroll
            for (int off = 1; off < 16; off <<= 1)
                rs += __shfl_xor(rs, off, 32);
            li[i] = li[i] * alpha + rs;
            #pragma unroll
            for (int nt = 0; nt < 4; ++nt) oacc[nt][i] *= alpha;
        }

        // O += P * V   (P re-read from LDS in A-fragment layout, V from LDS)
        v16bf af[2];
        #pragma unroll
        for (int s = 0; s < 2; ++s) {
            const unsigned short* base = Pw + lm * 64 + s * 32 + (lhalf ? 8 : 0);
            af[s] = load_frag(base, base + 16);
        }
        #pragma unroll
        for (int nt = 0; nt < 4; ++nt) {
            const unsigned short* vrow = &Vs[(nt * 16 + lm) * 64];
            #pragma unroll
            for (int s = 0; s < 2; ++s) {
                const unsigned short* base = vrow + s * 32 + (lhalf ? 8 : 0);
                v16bf vf = load_frag(base, base + 16);
                oacc[nt] = __builtin_amdgcn_wmma_f32_16x16x32_bf16(
                    false, af[s], false, vf, (short)0, oacc[nt], false, false);
            }
        }
    }

    // normalize + store O (bf16, row-major [b][l][h*64+d])
    #pragma unroll
    for (int i = 0; i < 8; ++i) {
        float inv = 1.0f / li[i];
        int row = q0 + i + lhalf * 8;
        unsigned short* dst = Ob + ((size_t)b * SEQ_L + row) * D_MODEL + h * HDIM;
        #pragma unroll
        for (int nt = 0; nt < 4; ++nt)
            dst[nt * 16 + lm] = f2bf(oacc[nt][i] * inv);
    }
}

// ---------------------------------------------------------------------------
extern "C" void kernel_launch(void* const* d_in, const int* in_sizes, int n_in,
                              void* d_out, int out_size, void* d_ws, size_t ws_size,
                              hipStream_t stream) {
    const float* x     = (const float*)d_in[0];
    const float* w_qkv = (const float*)d_in[1];
    const float* b_qkv = (const float*)d_in[2];
    const float* w_out = (const float*)d_in[3];
    const float* b_out = (const float*)d_in[4];
    float* out = (float*)d_out;

    const int n_x    = BATCH * SEQ_L * D_MODEL;     // 8388608
    const int n_wqkv = 3 * D_MODEL * D_MODEL;       // 3145728
    const int n_wout = D_MODEL * D_MODEL;           // 1048576

    unsigned short* p     = (unsigned short*)d_ws;
    unsigned short* xb    = p; p += n_x;
    unsigned short* wqkvb = p; p += n_wqkv;
    unsigned short* woutb = p; p += n_wout;
    unsigned short* Qb    = p; p += n_x;
    unsigned short* Kb    = p; p += n_x;
    unsigned short* Vt    = p; p += n_x;
    unsigned short* Ob    = p; p += n_x;

    cvt_f32_bf16<<<2048, 256, 0, stream>>>(x, xb, n_x);
    cvt_f32_bf16<<<1024, 256, 0, stream>>>(w_qkv, wqkvb, n_wqkv);
    cvt_f32_bf16<<<512,  256, 0, stream>>>(w_out, woutb, n_wout);

    // QKV projection + RoPE: M=8192, N=3072, K=1024
    gemm_bf16_wmma<0><<<dim3(3072 / 128, 8192 / 128), 256, 0, stream>>>(
        xb, wqkvb, b_qkv, 8192, 3072, 1024, Qb, Kb, Vt, nullptr);

    // Flash attention: 128 queries per block, one (b,h) per blockIdx.y
    attn_flash_wmma<<<dim3(SEQ_L / 128, BATCH * NHEADS), 256, 0, stream>>>(
        Qb, Kb, Vt, Ob);

    // Output projection: M=8192, N=1024, K=1024
    gemm_bf16_wmma<1><<<dim3(1024 / 128, 8192 / 128), 256, 0, stream>>>(
        Ob, woutb, b_out, 8192, 1024, 1024, nullptr, nullptr, nullptr, out);
}
